// MultiHeadAttentionBlock_35802847380028
// MI455X (gfx1250) — matrix operand order repaired and checked by reading, not compiled, not run on hardware
//
#include <hip/hip_runtime.h>
#include <hip/hip_bf16.h>
#include <math.h>

typedef __attribute__((ext_vector_type(16))) _Float16 v16h;
typedef __attribute__((ext_vector_type(8)))  float    v8f;

constexpr int D_MODEL   = 1024;
constexpr int NUM_HEADS = 16;
constexpr int DK        = 64;
constexpr int BATCH     = 2;
constexpr int SEQ       = 2048;
constexpr int MTOT      = BATCH * SEQ;   // 4096 rows

// ---------------------------------------------------------------------------
// Fragment-native LDS layouts (WMMA 16x16x32 f16; this code's contiguous k = 16h + s order).
//   A element (m,k): lane = (m&15) + 16*((k>>4)&1) ; elem = k&15
//   B element (k,n): lane = (n&15) + 16*((k>>4)&1) ; elem = k&15
// Each 32-lane x 16-half block = 512 halves; fragment load = one 32B v16h.
// ---------------------------------------------------------------------------
__device__ inline int a_frag_idx(int m, int k) {  // m in [0,64), k in [0,32)
  return (m >> 4) * 512 + ((m & 15) + 16 * ((k >> 4) & 1)) * 16 + (k & 15);
}
__device__ inline int b_frag_idx(int k, int n) {  // k in [0,32), n in [0,64)
  return (n >> 4) * 512 + ((n & 15) + 16 * ((k >> 4) & 1)) * 16 + (k & 15);
}

__device__ inline v16h frag_load(const _Float16* s, int half_off) {
  return *(const v16h*)(s + half_off);
}

__device__ inline v8f wmma32(v16h a, v16h b, v8f c) {
  return __builtin_amdgcn_wmma_f32_16x16x32_f16(false, a, false, b, (short)0, c,
                                                false, false);
}

// ---- CDNA5 async global->LDS copy (ASYNCcnt-tracked, ISA 15.18.3 op 98) ----
__device__ inline unsigned lds_off(const void* p) {
  // flat-LDS mapping: LDS address is the low 32 bits of the address (ISA 10.2)
  return (unsigned)(unsigned long long)p;
}
__device__ inline void async_copy_b128(unsigned lds_dst,
                                       unsigned long long gaddr) {
  asm volatile("global_load_async_to_lds_b128 %0, %1, off"
               :: "v"(lds_dst), "v"(gaddr) : "memory");
}
__device__ inline void wait_async0() {
  asm volatile("s_wait_asynccnt 0x0" ::: "memory");
}

union Pack4 { _Float16 h[4]; uint2 u; };
union Pack8 { _Float16 h[8]; uint4 u; };
union Pack2 { _Float16 h[2]; unsigned u; };

// ---------------------------------------------------------------------------
// GEMM: Out[M,N] = A[M,K] * W[N,K]^T + bias   (torch Linear semantics)
// Block = 128 threads (4 waves), tile 64x64, BK = 64 (2 WMMA k-steps/stage).
// ---------------------------------------------------------------------------
template <bool A_F16, bool OUT_F32>
__global__ __launch_bounds__(128) void gemm_xwT_kernel(
    const void* __restrict__ Aptr, const float* __restrict__ W,
    const float* __restrict__ bias, void* __restrict__ Outptr, int M, int N,
    int K) {
  // 2 k-steps x (4 blocks x 512 halves)
  __shared__ __align__(32) _Float16 As[4096];
  __shared__ __align__(32) _Float16 Bs[4096];

  const int tid    = threadIdx.x;
  const int lane   = tid & 31;
  const int wv     = tid >> 5;
  const int tile_n = blockIdx.x * 64;
  const int tile_m = blockIdx.y * 64;

  v8f acc[4];
#pragma unroll
  for (int nf = 0; nf < 4; ++nf)
#pragma unroll
    for (int r = 0; r < 8; ++r) acc[nf][r] = 0.f;

  for (int k0 = 0; k0 < K; k0 += 64) {
    __syncthreads();
    // ---- stage A tile (64 x 64) into fragment layout ----
    if constexpr (A_F16) {
      const _Float16* Ah = (const _Float16*)Aptr;
#pragma unroll
      for (int i = 0; i < 4; ++i) {
        int idx = tid + i * 128;  // 512 chunks of 8 halves (16B->16B copies)
        int row = idx >> 3, ch = idx & 7;
        const unsigned dst = lds_off(
            &As[(ch >> 2) * 2048 + a_frag_idx(row, (ch & 3) * 8)]);
        async_copy_b128(dst, (unsigned long long)(Ah +
                        (size_t)(tile_m + row) * K + k0 + ch * 8));
      }
    } else {
      const float* Af = (const float*)Aptr;
#pragma unroll
      for (int i = 0; i < 8; ++i) {
        int idx = tid + i * 128;  // 1024 float4
        int row = idx >> 4, c4 = idx & 15;
        const float4 v =
            *(const float4*)(Af + (size_t)(tile_m + row) * K + k0 + c4 * 4);
        Pack4 p;
        p.h[0] = (_Float16)v.x; p.h[1] = (_Float16)v.y;
        p.h[2] = (_Float16)v.z; p.h[3] = (_Float16)v.w;
        *(uint2*)(&As[(c4 >> 3) * 2048 + a_frag_idx(row, (c4 & 7) * 4)]) = p.u;
      }
    }
    // ---- stage B tile: element (k, n) = W[tile_n+n][k0+k] ----
#pragma unroll
    for (int i = 0; i < 8; ++i) {
      int idx = tid + i * 128;
      int n = idx >> 4, c4 = idx & 15;
      const float4 v =
          *(const float4*)(W + (size_t)(tile_n + n) * K + k0 + c4 * 4);
      Pack4 p;
      p.h[0] = (_Float16)v.x; p.h[1] = (_Float16)v.y;
      p.h[2] = (_Float16)v.z; p.h[3] = (_Float16)v.w;
      *(uint2*)(&Bs[(c4 >> 3) * 2048 + b_frag_idx((c4 & 7) * 4, n)]) = p.u;
    }
    // prefetch next W tile slice (global_prefetch_b8)
    if (k0 + 64 < K)
      __builtin_prefetch((const void*)(W + (size_t)(tile_n + (tid >> 1)) * K +
                                       k0 + 64), 0, 1);
    if constexpr (A_F16) wait_async0();
    __syncthreads();

#pragma unroll
    for (int ks = 0; ks < 2; ++ks) {
      const v16h a = frag_load(As, ks * 2048 + wv * 512 + lane * 16);
#pragma unroll
      for (int nf = 0; nf < 4; ++nf) {
        const v16h b = frag_load(Bs, ks * 2048 + nf * 512 + lane * 16);
        acc[nf] = wmma32(a, b, acc[nf]);
      }
    }
  }

  // ---- epilogue: bias + store (C/D layout: vgpr r, lane half -> row) ----
  const int nlo = lane & 15;
  const int mhi = (lane >> 4) * 8;
#pragma unroll
  for (int nf = 0; nf < 4; ++nf) {
    const int n    = tile_n + nf * 16 + nlo;
    const float bb = bias[n];
#pragma unroll
    for (int r = 0; r < 8; ++r) {
      const int m   = tile_m + wv * 16 + mhi + r;
      const float v = acc[nf][r] + bb;
      if constexpr (OUT_F32)
        ((float*)Outptr)[(size_t)m * N + n] = v;
      else
        ((_Float16*)Outptr)[(size_t)m * N + n] = (_Float16)v;
    }
  }
}

// ---------------------------------------------------------------------------
// Flash-style causal attention. One block per (query-tile of 64, b*H).
// Q/K/V are f16 in [B,S,D_MODEL]; head h = columns [h*64, h*64+64).
// Fragment-native LDS tiles; Q and K staged with async global->LDS copies.
// ---------------------------------------------------------------------------
__global__ __launch_bounds__(128) void attn_kernel(
    const _Float16* __restrict__ Q, const _Float16* __restrict__ Kd,
    const _Float16* __restrict__ Vd, _Float16* __restrict__ Ctx) {
  constexpr int SCS = 68;  // f32 score row stride
  __shared__ __align__(32) _Float16 Qf[4096];
  __shared__ __align__(32) _Float16 Kf[4096];
  __shared__ __align__(32) _Float16 Vf[4096];
  __shared__ __align__(32) _Float16 Pf[4096];
  __shared__ float Sc[64 * SCS];
  __shared__ float rowM[64], rowL[64], rowA[64];

  const int tid  = threadIdx.x;
  const int lane = tid & 31;
  const int wv   = tid >> 5;
  const int qt   = blockIdx.x;  // query tile index
  const int bh   = blockIdx.y;  // b * H + h
  const int b    = bh / NUM_HEADS;
  const int h    = bh % NUM_HEADS;

  const size_t headbase = (size_t)b * SEQ * D_MODEL + (size_t)h * DK;

  // ---- stage Q tile into A-fragment layout via async copies ----
#pragma unroll
  for (int i = 0; i < 4; ++i) {
    int idx = tid + i * 128;  // 64 rows x 8 chunks of 8 halves
    int row = idx >> 3, ch = idx & 7;
    const unsigned dst = lds_off(
        &Qf[(ch >> 2) * 2048 + a_frag_idx(row, (ch & 3) * 8)]);
    async_copy_b128(dst, (unsigned long long)(Q + headbase +
                    (size_t)(qt * 64 + row) * D_MODEL + ch * 8));
  }
  if (tid < 64) { rowM[tid] = -INFINITY; rowL[tid] = 0.f; }
  wait_async0();
  __syncthreads();

  const v16h aQ0 = frag_load(Qf, wv * 512 + lane * 16);
  const v16h aQ1 = frag_load(Qf, 2048 + wv * 512 + lane * 16);

  v8f O[4];
#pragma unroll
  for (int nf = 0; nf < 4; ++nf)
#pragma unroll
    for (int r = 0; r < 8; ++r) O[nf][r] = 0.f;

  const int nlo = lane & 15;
  const int mhi = (lane >> 4) * 8;

  for (int j = 0; j <= qt; ++j) {
    __syncthreads();  // protect K/V/P reuse from previous iteration

    // ---- stage K tile (B-layout, n = key, k = feature): async copies ----
#pragma unroll
    for (int i = 0; i < 4; ++i) {
      int idx = tid + i * 128;
      int key = idx >> 3, ch = idx & 7;
      const unsigned dst = lds_off(
          &Kf[(ch >> 2) * 2048 + b_frag_idx((ch & 3) * 8, key)]);
      async_copy_b128(dst, (unsigned long long)(Kd + headbase +
                      (size_t)(j * 64 + key) * D_MODEL + ch * 8));
    }
    // ---- stage V tile (B-layout, n = feature, k = key): packed stores ----
    // overlaps with the in-flight async K copies
#pragma unroll
    for (int i = 0; i < 2; ++i) {
      int u  = tid + i * 128;        // 256 units = 32 key-pairs x 8 f-chunks
      int kp = u >> 3, fc = u & 7;
      int kk = kp * 2, f0 = fc * 8;
      Pack8 pa, pb;
      pa.u = *(const uint4*)(Vd + headbase +
                             (size_t)(j * 64 + kk) * D_MODEL + f0);
      pb.u = *(const uint4*)(Vd + headbase +
                             (size_t)(j * 64 + kk + 1) * D_MODEL + f0);
      const int ks = kk >> 5, kr = kk & 31;
#pragma unroll
      for (int t = 0; t < 8; ++t) {
        Pack2 pr;
        pr.h[0] = pa.h[t];
        pr.h[1] = pb.h[t];
        *(unsigned*)(&Vf[ks * 2048 + b_frag_idx(kr, f0 + t)]) = pr.u;
      }
    }
    wait_async0();
    __syncthreads();

    // ---- scores S = Q * K^T ----
    v8f sc[4];
#pragma unroll
    for (int nf = 0; nf < 4; ++nf) {
#pragma unroll
      for (int r = 0; r < 8; ++r) sc[nf][r] = 0.f;
      const v16h b0 = frag_load(Kf, nf * 512 + lane * 16);
      const v16h b1 = frag_load(Kf, 2048 + nf * 512 + lane * 16);
      sc[nf] = wmma32(aQ0, b0, sc[nf]);
      sc[nf] = wmma32(aQ1, b1, sc[nf]);
    }
    // write scaled scores (scale = 1/sqrt(64) = 0.125)
#pragma unroll
    for (int nf = 0; nf < 4; ++nf)
#pragma unroll
      for (int r = 0; r < 8; ++r)
        Sc[(wv * 16 + mhi + r) * SCS + nf * 16 + nlo] = sc[nf][r] * 0.125f;
    __syncthreads();

    // ---- online softmax per row; P written in A-fragment layout ----
    if (tid < 64) {
      const int row   = tid;
      const int valid = (j == qt) ? (row + 1) : 64;  // causal
      const float mprev = rowM[row];
      float mt = -INFINITY;
      for (int c = 0; c < valid; ++c) mt = fmaxf(mt, Sc[row * SCS + c]);
      const float mnew  = fmaxf(mprev, mt);
      const float alpha = __expf(mprev - mnew);
      float lsum = 0.f;
#pragma unroll
      for (int g = 0; g < 8; ++g) {  // 8 groups of 8 columns
        Pack8 p;
#pragma unroll
        for (int jj = 0; jj < 8; ++jj) {
          const int c  = g * 8 + jj;
          const float pv =
              (c < valid) ? __expf(Sc[row * SCS + c] - mnew) : 0.f;
          lsum += pv;
          p.h[jj] = (_Float16)pv;
        }
        *(uint4*)(&Pf[(g >> 2) * 2048 + a_frag_idx(row, (g & 3) * 8)]) = p.u;
      }
      rowA[row] = alpha;
      rowM[row] = mnew;
      rowL[row] = alpha * rowL[row] + lsum;
    }
    __syncthreads();

    // ---- rescale O, then O += P * V ----
#pragma unroll
    for (int nf = 0; nf < 4; ++nf)
#pragma unroll
      for (int r = 0; r < 8; ++r) O[nf][r] *= rowA[wv * 16 + mhi + r];

    const v16h aP0 = frag_load(Pf, wv * 512 + lane * 16);
    const v16h aP1 = frag_load(Pf, 2048 + wv * 512 + lane * 16);
#pragma unroll
    for (int nf = 0; nf < 4; ++nf) {
      const v16h b0 = frag_load(Vf, nf * 512 + lane * 16);
      const v16h b1 = frag_load(Vf, 2048 + nf * 512 + lane * 16);
      O[nf] = wmma32(aP0, b0, O[nf]);
      O[nf] = wmma32(aP1, b1, O[nf]);
    }
  }
  __syncthreads();

  // ---- normalize and store context (f16 workspace) ----
#pragma unroll
  for (int nf = 0; nf < 4; ++nf) {
#pragma unroll
    for (int r = 0; r < 8; ++r) {
      const int row   = wv * 16 + mhi + r;
      const float inv = 1.f / rowL[row];
      const int q     = qt * 64 + row;
      Ctx[headbase + (size_t)q * D_MODEL + nf * 16 + nlo] =
          (_Float16)(O[nf][r] * inv);
    }
  }
}

// ---------------------------------------------------------------------------
extern "C" void kernel_launch(void* const* d_in, const int* in_sizes, int n_in,
                              void* d_out, int out_size, void* d_ws,
                              size_t ws_size, hipStream_t stream) {
  const float* q  = (const float*)d_in[0];
  const float* k  = (const float*)d_in[1];
  const float* v  = (const float*)d_in[2];
  const float* Wq = (const float*)d_in[3];
  const float* bq = (const float*)d_in[4];
  const float* Wk = (const float*)d_in[5];
  const float* bk = (const float*)d_in[6];
  const float* Wv = (const float*)d_in[7];
  const float* bv = (const float*)d_in[8];
  const float* Wo = (const float*)d_in[9];
  const float* bo = (const float*)d_in[10];
  float* out = (float*)d_out;

  const size_t seg = (size_t)MTOT * D_MODEL * sizeof(_Float16);  // 8 MB
  _Float16* Qw = (_Float16*)d_ws;
  _Float16* Kw = (_Float16*)((char*)d_ws + seg);
  _Float16* Vw = (_Float16*)((char*)d_ws + 2 * seg);
  _Float16* Cw = (_Float16*)((char*)d_ws + 3 * seg);

  dim3 blk(128);
  dim3 ggrid(D_MODEL / 64, MTOT / 64);  // (16, 64)

  hipLaunchKernelGGL((gemm_xwT_kernel<false, false>), ggrid, blk, 0, stream,
                     (const void*)q, Wq, bq, (void*)Qw, MTOT, D_MODEL, D_MODEL);
  hipLaunchKernelGGL((gemm_xwT_kernel<false, false>), ggrid, blk, 0, stream,
                     (const void*)k, Wk, bk, (void*)Kw, MTOT, D_MODEL, D_MODEL);
  hipLaunchKernelGGL((gemm_xwT_kernel<false, false>), ggrid, blk, 0, stream,
                     (const void*)v, Wv, bv, (void*)Vw, MTOT, D_MODEL, D_MODEL);

  dim3 agrid(SEQ / 64, BATCH * NUM_HEADS);  // (32, 32)
  hipLaunchKernelGGL(attn_kernel, agrid, blk, 0, stream, Qw, Kw, Vw, Cw);

  hipLaunchKernelGGL((gemm_xwT_kernel<true, true>), ggrid, blk, 0, stream,
                     (const void*)Cw, Wo, bo, (void*)out, MTOT, D_MODEL,
                     D_MODEL);
}
